// binary_tree_lstm_30073361007011
// MI455X (gfx1250) — compile-verified, matrix operand order checked by reading, not run on hardware
//
#include <hip/hip_runtime.h>

// ---------------- CDNA5 WMMA types ----------------
typedef __attribute__((ext_vector_type(16))) __bf16 v16bf;
typedef __attribute__((ext_vector_type(8)))  float  v8f;
typedef __attribute__((ext_vector_type(4)))  unsigned int v4u;

union FragU { v16bf v; v4u q[2]; };

__device__ __forceinline__ v8f wmma_bf16(v16bf a, v16bf b, v8f c) {
    // D = A(16x32 bf16) x B(32x16 bf16) + C(16x16 f32)
    return __builtin_amdgcn_wmma_f32_16x16x32_bf16(
        /*neg_a=*/false, a, /*neg_b=*/false, b,
        /*c_mod=*/(short)0, c, /*reuse_a=*/false, /*reuse_b=*/false);
}

__device__ __forceinline__ unsigned short f2bf(float f) {
    union { float f; unsigned int u; } x; x.f = f;
    unsigned int u = x.u;
    unsigned int r = (u + 0x7FFFu + ((u >> 16) & 1u)) >> 16;   // RNE
    return (unsigned short)r;
}

__device__ __forceinline__ float sigf(float x) {
    return 1.0f / (1.0f + __expf(-x));
}

// Problem dims
#define BATCH 512
#define LEN   256
#define INP   300
#define INPAD 320
#define HID   256
#define NCLS  5
#define NPOS  511   // 256+128+...+1

// ---------------- Pack kernels (fp32 -> bf16, WMMA-fragment layouts) ----------------

// Pad embeddings (B*L,300) f32 -> (B*L,320) bf16, zeros beyond 300.
__global__ void pack_emb_kernel(const float* __restrict__ emb,
                                unsigned short* __restrict__ dst) {
    long token = blockIdx.x;
    int k = threadIdx.x;                 // 0..319
    float v = (k < INP) ? emb[token * INP + k] : 0.0f;
    dst[token * INPAD + k] = f2bf(v);
}

// Fragment-elem mapping shared by the pack kernels:
//   lane<16  holds K offsets {0..7, 16..23}; lane>=16 holds {8..15, 24..31}
__device__ __forceinline__ int frag_klocal(int lane, int e) {
    return (e < 8 ? e : e + 8) + ((lane >= 16) ? 8 : 0);
}

// W_nl (10,256,256): virtual W (512 x 1280), K<256 from W_nl[2g], else W_nl[2g+1].
// Fragment layout: [((g*16+ht)*16+kt)*512 + lane*16 + e]
__global__ void pack_wnl_kernel(const float* __restrict__ w,
                                unsigned short* __restrict__ dst) {
    int idx = blockIdx.x * 256 + threadIdx.x;       // < 5*16*16*512 = 655360
    int e    = idx & 15;
    int lane = (idx >> 4)  & 31;
    int kt   = (idx >> 9)  & 15;
    int ht   = (idx >> 13) & 15;
    int g    =  idx >> 17;
    int k   = kt * 32 + frag_klocal(lane, e);
    int col = ht * 16 + (lane & 15);
    float v;
    if (k < 256) v = w[((long)(2 * g)     * 256 + k)        * 256 + col];
    else         v = w[((long)(2 * g + 1) * 256 + (k - 256)) * 256 + col];
    dst[idx] = f2bf(v);
}

// W_leaf (3,300,256): virtual W (320 x 768), zero-padded K>=300.
// Layout: [((g*16+ht)*10+kt)*512 + lane*16 + e]
__global__ void pack_wleaf_kernel(const float* __restrict__ w,
                                  unsigned short* __restrict__ dst) {
    int idx = blockIdx.x * 256 + threadIdx.x;       // < 3*16*10*512 = 245760
    int e    = idx & 15;
    int lane = (idx >> 4) & 31;
    int t    = idx >> 9;                            // < 480
    int kt   = t % 10; t /= 10;
    int ht   = t & 15;
    int g    = t >> 4;
    int k   = kt * 32 + frag_klocal(lane, e);
    int col = ht * 16 + (lane & 15);
    float v = (k < INP) ? w[((long)g * INP + k) * 256 + col] : 0.0f;
    dst[idx] = f2bf(v);
}

// W_out (256,5): padded to 256 x 16 (cols 5..15 zero). Layout: [kt*512 + lane*16 + e]
__global__ void pack_wout_kernel(const float* __restrict__ w,
                                 unsigned short* __restrict__ dst) {
    int idx = blockIdx.x * 256 + threadIdx.x;       // < 8*512 = 4096
    int e    = idx & 15;
    int lane = (idx >> 4) & 31;
    int kt   = idx >> 9;                            // 0..7
    int k   = kt * 32 + frag_klocal(lane, e);
    int col = lane & 15;
    float v = (col < NCLS) ? w[k * NCLS + col] : 0.0f;
    dst[idx] = f2bf(v);
}

// ---------------- Leaf kernel: pre = emb @ W_leaf + b_leaf; c,h ----------------
// 32 output rows (2 A-fragments) per wave; each B fragment reused twice.
// grid: (tokens/32, 16 hid-tiles), block: 32 (one wave).
__global__ __launch_bounds__(32) void leaf_kernel(
    const unsigned short* __restrict__ embp,     // (B*L, 320) bf16
    const unsigned short* __restrict__ wpack,    // leaf fragments
    const float* __restrict__ b_leaf,            // (3,256)
    float* __restrict__ c_out,                   // (B*L, 256) f32
    unsigned short* h_all)                       // (B, 511, 256) bf16
{
    const int lane  = threadIdx.x & 31;
    const int mbase = blockIdx.x << 5;
    const int ht    = blockIdx.y;
    const int ksel  = (lane >> 4) << 3;

    const unsigned short* ap0 = embp + (long)(mbase + (lane & 15))      * INPAD + ksel;
    const unsigned short* ap1 = embp + (long)(mbase + 16 + (lane & 15)) * INPAD + ksel;
    const unsigned short* wb  = wpack + (long)ht * (10 * 512) + lane * 16;

    const v8f zero = {0.f,0.f,0.f,0.f,0.f,0.f,0.f,0.f};
    v8f acc[2][3];
    #pragma unroll
    for (int s = 0; s < 2; ++s)
        #pragma unroll
        for (int g = 0; g < 3; ++g) acc[s][g] = zero;

    #pragma unroll
    for (int kt = 0; kt < 10; ++kt) {
        FragU fa0, fa1;
        fa0.q[0] = *(const v4u*)(ap0 + kt * 32);
        fa0.q[1] = *(const v4u*)(ap0 + kt * 32 + 16);
        fa1.q[0] = *(const v4u*)(ap1 + kt * 32);
        fa1.q[1] = *(const v4u*)(ap1 + kt * 32 + 16);
        #pragma unroll
        for (int g = 0; g < 3; ++g) {
            const unsigned short* wp = wb + (long)g * (16 * 10 * 512) + kt * 512;
            FragU fb;
            fb.q[0] = *(const v4u*)(wp);
            fb.q[1] = *(const v4u*)(wp + 8);
            acc[0][g] = wmma_bf16(fa0.v, fb.v, acc[0][g]);
            acc[1][g] = wmma_bf16(fa1.v, fb.v, acc[1][g]);
        }
    }

    const int col = lane & 15;
    const int hid = (ht << 4) + col;
    const float b0 = b_leaf[hid], b1 = b_leaf[256 + hid], b2 = b_leaf[512 + hid];

    #pragma unroll
    for (int s = 0; s < 2; ++s) {
        #pragma unroll
        for (int v = 0; v < 8; ++v) {
            int r = mbase + (s << 4) + v + ((lane >> 4) << 3);
            float ig = sigf(acc[s][0][v] + b0);
            float og = sigf(acc[s][1][v] + b1);
            float ug = tanhf(acc[s][2][v] + b2);
            float c  = ig * ug;
            float h  = og * sigf(c);
            c_out[(long)r * 256 + hid] = c;
            int b = r >> 8, l = r & 255;
            h_all[((long)(b * NPOS + l)) * 256 + hid] = f2bf(h);
        }
    }
}

// ---------------- Tree level kernel ----------------
// X row (K=512) = [h_left | h_right], contiguous since children are adjacent in h_all.
// 32 output rows per wave (2 A-fragments); each weight fragment reused twice.
// grid: (B*half/32, 16 hid-tiles), block: 32.
__global__ __launch_bounds__(32) void tree_level_kernel(
    const unsigned short* h_all_r,
    const float* __restrict__ c_in,
    float* __restrict__ c_out,
    unsigned short* h_all_w,
    const unsigned short* __restrict__ wpack,    // W_nl fragments (virtual 512x1280)
    const float* __restrict__ b_nl,              // (10,256)
    int halfShift, int off_in, int off_out)
{
    const int lane  = threadIdx.x & 31;
    const int mbase = blockIdx.x << 5;
    const int ht    = blockIdx.y;
    const int half  = 1 << halfShift;
    const int ksel  = (lane >> 4) << 3;

    int r0 = mbase + (lane & 15);
    int r1 = r0 + 16;
    const unsigned short* pl0 = h_all_r +
        ((long)((r0 >> halfShift) * NPOS + off_in + 2 * (r0 & (half - 1)))) * 256 + ksel;
    const unsigned short* pl1 = h_all_r +
        ((long)((r1 >> halfShift) * NPOS + off_in + 2 * (r1 & (half - 1)))) * 256 + ksel;
    const unsigned short* wb = wpack + (long)ht * (16 * 512) + lane * 16;

    const v8f zero = {0.f,0.f,0.f,0.f,0.f,0.f,0.f,0.f};
    v8f acc[2][5];
    #pragma unroll
    for (int s = 0; s < 2; ++s)
        #pragma unroll
        for (int g = 0; g < 5; ++g) acc[s][g] = zero;

    #pragma unroll
    for (int kt = 0; kt < 16; ++kt) {
        FragU fa0, fa1;
        fa0.q[0] = *(const v4u*)(pl0 + kt * 32);
        fa0.q[1] = *(const v4u*)(pl0 + kt * 32 + 16);
        fa1.q[0] = *(const v4u*)(pl1 + kt * 32);
        fa1.q[1] = *(const v4u*)(pl1 + kt * 32 + 16);
        #pragma unroll
        for (int g = 0; g < 5; ++g) {
            const unsigned short* wp = wb + (long)g * (16 * 16 * 512) + kt * 512;
            FragU fb;
            fb.q[0] = *(const v4u*)(wp);
            fb.q[1] = *(const v4u*)(wp + 8);
            acc[0][g] = wmma_bf16(fa0.v, fb.v, acc[0][g]);
            acc[1][g] = wmma_bf16(fa1.v, fb.v, acc[1][g]);
        }
    }

    const int col = lane & 15;
    const int hid = (ht << 4) + col;
    float bb[5];
    #pragma unroll
    for (int g = 0; g < 5; ++g)
        bb[g] = b_nl[(2 * g) * 256 + hid] + b_nl[(2 * g + 1) * 256 + hid];

    const long nin = (long)half << 1;
    #pragma unroll
    for (int s = 0; s < 2; ++s) {
        #pragma unroll
        for (int v = 0; v < 8; ++v) {
            int r = mbase + (s << 4) + v + ((lane >> 4) << 3);
            int b = r >> halfShift;
            int i = r & (half - 1);
            const float* cp = c_in + ((long)b * nin + 2 * i) * 256 + hid;
            float lc = cp[0];
            float rc = cp[256];
            float ig = sigf(acc[s][0][v] + bb[0]);
            float fl = sigf(acc[s][1][v] + bb[1]);
            float fr = sigf(acc[s][2][v] + bb[2]);
            float og = sigf(acc[s][3][v] + bb[3]);
            float ug = tanhf(acc[s][4][v] + bb[4]);
            float c  = ig * ug + fl * lc + fr * rc;
            float h  = og * tanhf(c);
            c_out[((long)b * half + i) * 256 + hid] = c;
            h_all_w[((long)(b * NPOS + off_out + i)) * 256 + hid] = f2bf(h);
        }
    }
}

// ---------------- Logits: h_all (B*511,256) @ W_out(256,16-padded) ----------------
// 32 rows per wave. grid: (B*511/32), block: 32.
__global__ __launch_bounds__(32) void logits_kernel(
    const unsigned short* __restrict__ h_all,
    const unsigned short* __restrict__ wpack,    // padded W_out fragments
    const float* __restrict__ b_out,
    float* __restrict__ out)
{
    const int lane = threadIdx.x & 31;
    const long tile = blockIdx.x;
    const int ksel = (lane >> 4) << 3;
    const unsigned short* ap0 = h_all + (tile * 32 + (lane & 15))      * 256 + ksel;
    const unsigned short* ap1 = h_all + (tile * 32 + 16 + (lane & 15)) * 256 + ksel;
    const unsigned short* wb = wpack + lane * 16;

    const v8f zero = {0.f,0.f,0.f,0.f,0.f,0.f,0.f,0.f};
    v8f acc0 = zero, acc1 = zero;
    #pragma unroll
    for (int kt = 0; kt < 8; ++kt) {
        FragU fa0, fa1, fb;
        fa0.q[0] = *(const v4u*)(ap0 + kt * 32);
        fa0.q[1] = *(const v4u*)(ap0 + kt * 32 + 16);
        fa1.q[0] = *(const v4u*)(ap1 + kt * 32);
        fa1.q[1] = *(const v4u*)(ap1 + kt * 32 + 16);
        const unsigned short* wp = wb + kt * 512;
        fb.q[0] = *(const v4u*)(wp);
        fb.q[1] = *(const v4u*)(wp + 8);
        acc0 = wmma_bf16(fa0.v, fb.v, acc0);
        acc1 = wmma_bf16(fa1.v, fb.v, acc1);
    }

    const int col = lane & 15;
    if (col < NCLS) {
        float bo = b_out[col];
        #pragma unroll
        for (int v = 0; v < 8; ++v) {
            long r0 = tile * 32 + v + ((lane >> 4) << 3);
            out[r0 * NCLS + col] = acc0[v] + bo;
            out[(r0 + 16) * NCLS + col] = acc1[v] + bo;
        }
    }
}

// ---------------- Launch ----------------
extern "C" void kernel_launch(void* const* d_in, const int* in_sizes, int n_in,
                              void* d_out, int out_size, void* d_ws, size_t ws_size,
                              hipStream_t stream) {
    (void)in_sizes; (void)n_in; (void)out_size; (void)ws_size;
    const float* emb    = (const float*)d_in[0];
    const float* W_leaf = (const float*)d_in[1];
    const float* b_leaf = (const float*)d_in[2];
    const float* W_nl   = (const float*)d_in[3];
    const float* b_nl   = (const float*)d_in[4];
    const float* W_out  = (const float*)d_in[5];
    const float* b_out  = (const float*)d_in[6];
    float* out = (float*)d_out;

    char* ws = (char*)d_ws;
    // Workspace carve-out (all offsets 256B aligned)
    unsigned short* wpnl   = (unsigned short*)(ws + 0);          //  1,310,720 B
    unsigned short* wpleaf = (unsigned short*)(ws + 1310720);    //    491,520 B
    unsigned short* wpout  = (unsigned short*)(ws + 1802240);    //      8,192 B
    unsigned short* embp   = (unsigned short*)(ws + 1810432);    // 83,886,080 B
    unsigned short* h_all  = (unsigned short*)(ws + 85696512);   // 133,955,584 B
    float*          cA     = (float*)(ws + 219652096);           // 134,217,728 B
    float*          cB     = (float*)(ws + 353869824);           //  67,108,864 B

    // Weight / input packing (bf16 fragment layouts)
    hipLaunchKernelGGL(pack_emb_kernel,   dim3(BATCH * LEN), dim3(INPAD), 0, stream, emb, embp);
    hipLaunchKernelGGL(pack_wnl_kernel,   dim3(655360 / 256), dim3(256),  0, stream, W_nl, wpnl);
    hipLaunchKernelGGL(pack_wleaf_kernel, dim3(245760 / 256), dim3(256),  0, stream, W_leaf, wpleaf);
    hipLaunchKernelGGL(pack_wout_kernel,  dim3(4096 / 256),   dim3(256),  0, stream, W_out, wpout);

    // Leaf level: 131072 tokens, 32 rows per wave
    hipLaunchKernelGGL(leaf_kernel, dim3((BATCH * LEN) / 32, 16), dim3(32), 0, stream,
                       embp, wpleaf, b_leaf, cA, h_all);

    // Tree levels
    static const int offs[9] = {0, 256, 384, 448, 480, 496, 504, 508, 510};
    const float* ci = cA;
    for (int lvl = 1; lvl <= 8; ++lvl) {
        int halfShift = 8 - lvl;
        int half = 1 << halfShift;
        float* co = (lvl & 1) ? cB : cA;
        hipLaunchKernelGGL(tree_level_kernel,
                           dim3((BATCH * half) / 32, 16), dim3(32), 0, stream,
                           h_all, ci, co, h_all, wpnl, b_nl,
                           halfShift, offs[lvl - 1], offs[lvl]);
        ci = co;
    }

    // Logits over all 511 positions per batch
    hipLaunchKernelGGL(logits_kernel, dim3((BATCH * NPOS) / 32), dim3(32), 0, stream,
                       h_all, wpout, b_out, out);
}
